// CustomLSTM_29549374996871
// MI455X (gfx1250) — compile-verified, matrix operand order checked by reading, not run on hardware
//
#include <hip/hip_runtime.h>
#include <hip/hip_bf16.h>

typedef __bf16  v16bf __attribute__((ext_vector_type(16)));
typedef __bf16  v8bf  __attribute__((ext_vector_type(8)));
typedef float   v8f   __attribute__((ext_vector_type(8)));

#define T_STEPS 256
#define BATCH   256
#define IDIM    128
#define SDIM    256
#define KTILES  12            // 4 x-ktiles (K=0..127) + 8 s-ktiles (K=128..383)
#define NBLOCKS 16            // one block per 16 batch rows; recurrence is batch-local
#define NTHREADS 512          // 16 waves; wave wid owns col-tile C = wid

// workspace layout (bytes); total ~790 KB
#define BB_OFF    0                       // concat bias f32[1024]
#define PACK_OFF  4096                    // packed weights bf16[384*1024]

#define PACK_ELEMS  (384u * 1024u)        // 393216
#define BIAS_ELEMS  1024u
#define PREP_TOTAL  (PACK_ELEMS + BIAS_ELEMS)

// LDS geometry (halves): padded row strides keep 16B alignment + spread banks
#define S_STRIDE   264                    // 256 cols + 8 pad  (528 B rows)
#define X_STRIDE   136                    // 128 cols + 8 pad  (272 B rows)

// ---------------------------------------------------------------------------
// Prep: pack [W;U] (logical f32 [384 x 1024], gate-major cols i|f|o|g) into a
// bf16 layout where a wave's B-operand (32x16 KxN tile) is 32 lanes x 32
// contiguous bytes.  B layout mirrors the ISA A layout:
//   lane = n + 16*((k>>3)&1),  half = (k&7) + 8*(k>>4)
// ---------------------------------------------------------------------------
__global__ void lstm_prep(
    const float* __restrict__ Wi, const float* __restrict__ Ui, const float* __restrict__ Bi,
    const float* __restrict__ Wf, const float* __restrict__ Uf, const float* __restrict__ Bf,
    const float* __restrict__ Wo, const float* __restrict__ Uo, const float* __restrict__ Bo,
    const float* __restrict__ Wg, const float* __restrict__ Ug, const float* __restrict__ Bg,
    unsigned short* __restrict__ packU16, float* __restrict__ Bb)
{
  unsigned gid = blockIdx.x * blockDim.x + threadIdx.x;
  if (gid < PACK_ELEMS) {
    __bf16* pack = reinterpret_cast<__bf16*>(packU16);
    unsigned frag  = gid >> 9;            // 512 elements per fragment
    unsigned rem   = gid & 511u;
    unsigned lane  = rem >> 4;
    unsigned h     = rem & 15u;
    unsigned ntile = frag / KTILES;
    unsigned ktile = frag - ntile * KTILES;
    unsigned k = (h & 7u) | ((lane >> 4) << 3) | ((h >> 3) << 4);   // 0..31
    unsigned n = lane & 15u;
    unsigned kk = ktile * 32u + k;        // 0..383  (0..127 = W rows, rest U rows)
    unsigned nn = ntile * 16u + n;        // 0..1023
    unsigned gate = nn >> 8, col = nn & 255u;
    const float* Wp[4] = {Wi, Wf, Wo, Wg};
    const float* Up[4] = {Ui, Uf, Uo, Ug};
    float v = (kk < 128u) ? Wp[gate][kk * 256u + col]
                          : Up[gate][(kk - 128u) * 256u + col];
    pack[frag * 512u + lane * 16u + h] = (__bf16)v;
  } else if (gid < PACK_ELEMS + BIAS_ELEMS) {
    unsigned j = gid - PACK_ELEMS;
    const float* Bp[4] = {Bi, Bf, Bo, Bg};
    Bb[j] = Bp[j >> 8][j & 255u];
  }
}

// ---------------------------------------------------------------------------
// A-operand fragment (16x32 bf16) from LDS, ISA layout:
//   lanes 0-15 : row M=lane,    halves 0-7 = K[b..b+7],   8-15 = K[b+16..b+23]
//   lanes 16-31: row M=lane-16, halves 0-7 = K[b+8..b+15],8-15 = K[b+24..b+31]
// ---------------------------------------------------------------------------
__device__ __forceinline__ v16bf lds_frag_a(const __bf16* base, int rowStride,
                                            int kBase, int lane) {
  int row = lane & 15;
  int kg  = lane >> 4;
  const __bf16* p = base + row * rowStride + kBase + kg * 8;
  union { v16bf v; v8bf h[2]; } u;
  u.h[0] = *(const v8bf*)(p);        // ds_read_b128
  u.h[1] = *(const v8bf*)(p + 16);   // ds_read_b128
  return u.v;
}

// B fragment streamed from the L2-resident pack: plain global_load_b128 pair
// (clause-able, normal scope, waits by real dependence).  The caller passes a
// base made opaque per timestep so these cannot be hoisted across the t-loop.
__device__ __forceinline__ v16bf ld_frag_b(const __bf16* packt, int ntile,
                                           int ktile, int lane) {
  const v8bf* p =
      (const v8bf*)(packt + (((ntile * KTILES) + ktile) * 32 + lane) * 16);
  union { v16bf v; v8bf h[2]; } u;
  u.h[0] = p[0];
  u.h[1] = p[1];
  return u.v;
}

__device__ __forceinline__ float sigmoidf_(float z) {
  return 1.0f / (1.0f + __expf(-z));
}

// ---------------------------------------------------------------------------
// Persistent recurrent kernel.  The LSTM recurrence couples hidden units
// within one batch row only, so each block owns 16 batch rows and runs all
// 512 steps with the state resident in its own LDS: no global state, no
// device-wide sync — only the workgroup barrier between steps.
//   16 waves; wave wid owns col-tile C = wid (c tile lives in VGPRs).
// ---------------------------------------------------------------------------
__global__ __launch_bounds__(NTHREADS)
void lstm_persistent(
    const float* __restrict__ x,
    const unsigned short* __restrict__ packU16,
    const float* __restrict__ Bb,
    float* __restrict__ out)
{
  const __bf16* pack = reinterpret_cast<const __bf16*>(packU16);

  __shared__ __bf16 lds_s[2 * 16 * S_STRIDE];   // double-buffered state rows
  __shared__ __bf16 lds_x[16 * X_STRIDE];       // staged x rows (encode)

  const int tid  = threadIdx.x;
  const int lane = tid & 31;
  const int C    = tid >> 5;                    // wave id == col tile 0..15
  const int R    = blockIdx.x;                  // 16 batch rows R*16..R*16+15
  const int nl   = lane & 15;
  const int hh   = lane >> 4;
  const int col  = C * 16 + nl;

  // zero both state buffers (and pads) — s0 = 0
  {
    unsigned* z = (unsigned*)lds_s;
    for (int i = tid; i < (2 * 16 * S_STRIDE) / 2; i += NTHREADS) z[i] = 0u;
  }

  // per-lane gate biases (column-invariant across the 8 accum VGPRs)
  const float bb0 = Bb[0 * 256 + col];
  const float bb1 = Bb[1 * 256 + col];
  const float bb2 = Bb[2 * 256 + col];
  const float bb3 = Bb[3 * 256 + col];

  float cc[8];
  #pragma unroll
  for (int r = 0; r < 8; ++r) cc[r] = 0.0f;

  __syncthreads();

  for (int t = 0; t < 2 * T_STEPS; ++t) {
    const bool enc = (t < T_STEPS);
    const __bf16* scur = lds_s + (t & 1) * (16 * S_STRIDE);
    __bf16*       snxt = lds_s + ((t + 1) & 1) * (16 * S_STRIDE);

    // Opaque zero regenerated every step: the compiler cannot prove the pack
    // base is loop-invariant, so the weight loads stay inside the loop as
    // ordinary pipelined global loads (no hoist -> no spill, no volatile ->
    // no SYS-scope serialized flat loads).
    int opaque0;
    asm volatile("s_mov_b32 %0, 0" : "=s"(opaque0));
    const __bf16* packt = pack + opaque0;

    // ---- stage x rows for this step (encode only) + prefetch next step ----
    if (enc) {
      {
        int ch = tid;                            // 16 rows x 32 float4 chunks
        int r  = ch >> 5, c4 = ch & 31;
        const float* gp =
            x + ((size_t)t * BATCH + R * 16 + r) * IDIM + c4 * 4;
        const float4 v = *(const float4*)gp;
        __bf16* p = lds_x + r * X_STRIDE + c4 * 4;
        p[0] = (__bf16)v.x; p[1] = (__bf16)v.y;
        p[2] = (__bf16)v.z; p[3] = (__bf16)v.w;
        if (t + 1 < T_STEPS)
          __builtin_prefetch(gp + (size_t)BATCH * IDIM, 0, 1);  // global_prefetch_b8
      }
      __syncthreads();
    }

    // ---- z tile = concat(x_t, s) @ WU + b  (bias folded into accum init) ----
    v8f a0, a1, a2, a3;
    #pragma unroll
    for (int r = 0; r < 8; ++r) {
      a0[r] = bb0; a1[r] = bb1; a2[r] = bb2; a3[r] = bb3;
    }

    // depth-1 software pipeline: A (LDS) and 4 B frags (L2) for k+1 in
    // flight while the 4 WMMAs of k execute.
    v16bf acur = lds_frag_a(scur, S_STRIDE, 0, lane);
    v16bf nb0 = ld_frag_b(packt,  0 + C, 4, lane);
    v16bf nb1 = ld_frag_b(packt, 16 + C, 4, lane);
    v16bf nb2 = ld_frag_b(packt, 32 + C, 4, lane);
    v16bf nb3 = ld_frag_b(packt, 48 + C, 4, lane);

    #pragma unroll
    for (int kt = 0; kt < 8; ++kt) {             // recurrent part (K=256)
      v16bf a = acur, c0 = nb0, c1 = nb1, c2 = nb2, c3 = nb3;
      if (kt < 7) {
        acur = lds_frag_a(scur, S_STRIDE, (kt + 1) * 32, lane);
        nb0 = ld_frag_b(packt,  0 + C, 5 + kt, lane);
        nb1 = ld_frag_b(packt, 16 + C, 5 + kt, lane);
        nb2 = ld_frag_b(packt, 32 + C, 5 + kt, lane);
        nb3 = ld_frag_b(packt, 48 + C, 5 + kt, lane);
      } else if (enc) {                          // bridge into the x part
        acur = lds_frag_a(lds_x, X_STRIDE, 0, lane);
        nb0 = ld_frag_b(packt,  0 + C, 0, lane);
        nb1 = ld_frag_b(packt, 16 + C, 0, lane);
        nb2 = ld_frag_b(packt, 32 + C, 0, lane);
        nb3 = ld_frag_b(packt, 48 + C, 0, lane);
      }
      a0 = __builtin_amdgcn_wmma_f32_16x16x32_bf16(false, a, false, c0, (short)0, a0, false, false);
      a1 = __builtin_amdgcn_wmma_f32_16x16x32_bf16(false, a, false, c1, (short)0, a1, false, false);
      a2 = __builtin_amdgcn_wmma_f32_16x16x32_bf16(false, a, false, c2, (short)0, a2, false, false);
      a3 = __builtin_amdgcn_wmma_f32_16x16x32_bf16(false, a, false, c3, (short)0, a3, false, false);
    }
    if (enc) {
      #pragma unroll
      for (int kt = 0; kt < 4; ++kt) {           // input part (K=128); decode x==0
        v16bf a = acur, c0 = nb0, c1 = nb1, c2 = nb2, c3 = nb3;
        if (kt < 3) {
          acur = lds_frag_a(lds_x, X_STRIDE, (kt + 1) * 32, lane);
          nb0 = ld_frag_b(packt,  0 + C, kt + 1, lane);
          nb1 = ld_frag_b(packt, 16 + C, kt + 1, lane);
          nb2 = ld_frag_b(packt, 32 + C, kt + 1, lane);
          nb3 = ld_frag_b(packt, 48 + C, kt + 1, lane);
        }
        a0 = __builtin_amdgcn_wmma_f32_16x16x32_bf16(false, a, false, c0, (short)0, a0, false, false);
        a1 = __builtin_amdgcn_wmma_f32_16x16x32_bf16(false, a, false, c1, (short)0, a1, false, false);
        a2 = __builtin_amdgcn_wmma_f32_16x16x32_bf16(false, a, false, c2, (short)0, a2, false, false);
        a3 = __builtin_amdgcn_wmma_f32_16x16x32_bf16(false, a, false, c3, (short)0, a3, false, false);
      }
    }

    // ---- gates + state update (f32 VALU); accum VGPR r -> row M=r+8*hh ----
    #pragma unroll
    for (int r = 0; r < 8; ++r) {
      float ig = sigmoidf_(a0[r]);
      float fg = sigmoidf_(a1[r]);
      float og = sigmoidf_(a2[r]);
      float gg = tanhf(a3[r]);
      cc[r] = cc[r] * fg + gg * ig;
      float ss = tanhf(cc[r]) * og;
      int rowl = r + 8 * hh;                     // local row 0..15
      snxt[rowl * S_STRIDE + col] = (__bf16)ss;
      if (!enc) {
        float* op = out + (size_t)(t - T_STEPS) * (BATCH * SDIM)
                        + (R * 16 + rowl) * SDIM + col;
        __builtin_nontemporal_store(og, op);     // streaming 64MB, keep L2 clean
      }
    }

    __syncthreads();   // snxt complete; next step reads it as scur
  }
}

// ---------------------------------------------------------------------------
extern "C" void kernel_launch(void* const* d_in, const int* in_sizes, int n_in,
                              void* d_out, int out_size, void* d_ws, size_t ws_size,
                              hipStream_t stream) {
  const float* x  = (const float*)d_in[0];
  const float* Wi = (const float*)d_in[1],  *Ui = (const float*)d_in[2],  *Bi = (const float*)d_in[3];
  const float* Wf = (const float*)d_in[4],  *Uf = (const float*)d_in[5],  *Bf = (const float*)d_in[6];
  const float* Wo = (const float*)d_in[7],  *Uo = (const float*)d_in[8],  *Bo = (const float*)d_in[9];
  const float* Wg = (const float*)d_in[10], *Ug = (const float*)d_in[11], *Bg = (const float*)d_in[12];

  char* ws = (char*)d_ws;
  float*          Bb   = (float*)(ws + BB_OFF);
  unsigned short* pack = (unsigned short*)(ws + PACK_OFF);

  const unsigned prepBlocks = (PREP_TOTAL + 255u) / 256u;
  lstm_prep<<<prepBlocks, 256, 0, stream>>>(Wi, Ui, Bi, Wf, Uf, Bf,
                                            Wo, Uo, Bo, Wg, Ug, Bg,
                                            pack, Bb);
  lstm_persistent<<<NBLOCKS, NTHREADS, 0, stream>>>(x, pack, Bb, (float*)d_out);
}
